// InputEmbedder_84430467105277
// MI455X (gfx1250) — matrix operand order repaired and checked by reading, not compiled, not run on hardware
//
#include <hip/hip_runtime.h>
#include <hip/hip_bf16.h>

// Problem constants (match reference)
#define LSEQ  512
#define CM    256
#define CZ    128
#define NC    128
#define NBINS 65
#define KF_T  20   // target_feat channels
#define KF_M  21   // msa_feat channels

typedef __attribute__((ext_vector_type(16))) _Float16 v16h;
typedef __attribute__((ext_vector_type(8)))  float    v8f;
typedef __attribute__((ext_vector_type(4)))  float    v4f;

// ---------------------------------------------------------------------------
// Kernel 1: pair_rep [CZ, L, L] = a[c,i] + b[c,j] + w_p[c,bin(i,j)] + w_p_b[c]
// One block handles (c, 64-row chunk of i). Streaming NT float4 stores.
// ---------------------------------------------------------------------------
__global__ __launch_bounds__(256) void pair_kernel(
    const float* __restrict__ tf,     // [20, 512]
    const int*   __restrict__ ri,     // [512]
    const float* __restrict__ waw,    // [128, 20]
    const float* __restrict__ wab,    // [128]
    const float* __restrict__ wbw,    // [128, 20]
    const float* __restrict__ wbb,    // [128]
    const float* __restrict__ wpw,    // [128, 65]
    const float* __restrict__ wpb,    // [128]
    float* __restrict__ out)          // pair_rep base [128, 512, 512]
{
    const int c  = blockIdx.x >> 3;         // 0..127
    const int i0 = (blockIdx.x & 7) * 64;   // row chunk base
    const int t  = threadIdx.x;             // 0..255

    __shared__ float sa[64];
    __shared__ float sb[LSEQ];
    __shared__ float sp[NBINS];
    __shared__ int   sri[LSEQ];

    sri[t]       = ri[t];
    sri[t + 256] = ri[t + 256];

    // b[c, j] for j = t and t+256  (K=20 dot, inputs L2-resident)
    {
        float b0 = wbb[c], b1 = wbb[c];
        #pragma unroll
        for (int f = 0; f < KF_T; ++f) {
            const float w = wbw[c * KF_T + f];
            b0 += w * tf[f * LSEQ + t];
            b1 += w * tf[f * LSEQ + t + 256];
        }
        sb[t]       = b0;
        sb[t + 256] = b1;
    }
    // a[c, i0+t] for t < 64
    if (t < 64) {
        float a0 = wab[c];
        #pragma unroll
        for (int f = 0; f < KF_T; ++f)
            a0 += waw[c * KF_T + f] * tf[f * LSEQ + i0 + t];
        sa[t] = a0;
    }
    // fold w_p_b into the bin table
    if (t < NBINS) sp[t] = wpw[c * NBINS + t] + wpb[c];
    __syncthreads();

    float* __restrict__ orow = out + (size_t)c * LSEQ * LSEQ;
    const int jq   = (t & 127) * 4;  // float4 column
    const int isel = t >> 7;         // 0 or 1: which of the 2 rows

    for (int ii = 0; ii < 64; ii += 2) {
        const int   i   = i0 + ii + isel;
        const float ai  = sa[ii + isel];
        const int   rii = sri[i];
        v4f v;
        #pragma unroll
        for (int e = 0; e < 4; ++e) {
            const int j = jq + e;
            int d = rii - sri[j];
            d = (d < -32) ? -32 : (d > 32 ? 32 : d);
            v[e] = ai + sb[j] + sp[d + 32];
        }
        __builtin_nontemporal_store(v, (v4f*)(orow + (size_t)i * LSEQ + jq));
    }
}

// ---------------------------------------------------------------------------
// Kernel 2: msa_rep [CM, NC, L] via WMMA f32<-f16, K=21 padded into one
// 16x16x32 WMMA per tile. Row r = c*128 + k. Block: fixed (c, kTile),
// 8 waves each take a 64-column chunk (4 N-tiles -> 4 WMMAs, A reused).
// Accumulator seeded with fused bias w_clust_b[k,c] + m[c,col].
// K-padding via clamped-address loads + value select (uniform control flow,
// no exec-mask branches).
// ---------------------------------------------------------------------------
__global__ __launch_bounds__(256) void msa_kernel(
    const float* __restrict__ tf,     // [20, 512]
    const float* __restrict__ msa,    // [21, 512]
    const float* __restrict__ wcw,    // [128, 256, 21]  (k, c, f)
    const float* __restrict__ wcb,    // [128, 256]      (k, c)
    const float* __restrict__ wmw,    // [256, 20]
    const float* __restrict__ wmb,    // [256]
    float* __restrict__ out)          // msa_rep base [256, 128, 512]
{
    const int c      = blockIdx.x >> 3;   // 0..255 (CM index)
    const int kTile  = blockIdx.x & 7;    // 0..7   (16 k-rows each)
    const int wave   = threadIdx.x >> 5;  // 0..7 -> 64-col chunk
    const int lane   = threadIdx.x & 31;
    const int laneHi = lane >> 4;         // 0/1
    const int ln     = lane & 15;
    const int colBase = wave * 64;

    // ---- A fragment: 16x32 f16, M = ln, K striping per ISA 7.12.2 ----
    // lanes 0-15: h0..7 -> K=0..7,  h8..15 -> K=16..23
    // lanes16-31: h0..7 -> K=8..15, h8..15 -> K=24..31
    const int kRow = kTile * 16 + ln;
    const float* __restrict__ arow = wcw + ((size_t)kRow * CM + c) * KF_M;
    const int kb = laneHi * 8;
    v16h a;
    // First 8 K-slots: f = kb + h <= 15 < 21, always valid: unconditional.
    #pragma unroll
    for (int h = 0; h < 8; ++h) {
        a[h] = (_Float16)arow[kb + h];
    }
    // Second 8 K-slots: f = 16 + kb + h in 16..31; clamp address, select value.
    #pragma unroll
    for (int h = 0; h < 8; ++h) {
        const int f  = 16 + kb + h;
        const int fc = (f < KF_M) ? f : (KF_M - 1);
        const float x = arow[fc];                      // unconditional load
        a[8 + h] = (f < KF_M) ? (_Float16)x : (_Float16)0.f;  // VALU select
    }

    // per-output-row bias: k_v = kTile*16 + v + laneHi*8
    float cb[8];
    #pragma unroll
    for (int v = 0; v < 8; ++v)
        cb[v] = wcb[(size_t)(kTile * 16 + laneHi * 8 + v) * CM + c];

    // w_m row for this c (broadcast loads, L2-resident)
    float wm[KF_T];
    #pragma unroll
    for (int f = 0; f < KF_T; ++f) wm[f] = wmw[c * KF_T + f];
    const float wmbc = wmb[c];

    const int kb0 = laneHi * 8;  // B: lanes 0-15 K=0..7,16..23; lanes 16-31 K=8..15,24..31

    #pragma unroll
    for (int nt = 0; nt < 4; ++nt) {
        const int col = colBase + nt * 16 + ln;

        // ---- B fragment: 32x16 f16, N = ln ----
        // Clamp address, select value: uniform control flow, clause-able loads.
        v16h b;
        #pragma unroll
        for (int h = 0; h < 16; ++h) {
            const int f  = (h < 8) ? (kb0 + h) : (16 + kb0 + (h - 8));
            const int fc = (f < KF_M) ? f : (KF_M - 1);
            const float x = msa[fc * LSEQ + col];      // unconditional load
            b[h] = (f < KF_M) ? (_Float16)x : (_Float16)0.f;
        }

        // m[c, col] = w_m . target_feat[:, col] + w_m_b[c]
        float mval = wmbc;
        #pragma unroll
        for (int f = 0; f < KF_T; ++f) mval += wm[f] * tf[f * LSEQ + col];

        // seed accumulator with fused bias, one WMMA covers all of K
        v8f acc;
        #pragma unroll
        for (int v = 0; v < 8; ++v) acc[v] = cb[v] + mval;

        acc = __builtin_amdgcn_wmma_f32_16x16x32_f16(
            /*neg_a=*/false, a, /*neg_b=*/false, b,
            /*c_mod=*/(short)0, acc, /*reuse_a=*/false, /*reuse_b=*/false);

        // D layout: VGPR v -> M = v + laneHi*8, N = ln
        float* __restrict__ obase =
            out + ((size_t)c * NC + kTile * 16 + laneHi * 8) * LSEQ + col;
        #pragma unroll
        for (int v = 0; v < 8; ++v)
            __builtin_nontemporal_store(acc[v], obase + (size_t)v * LSEQ);
    }
}

// ---------------------------------------------------------------------------
extern "C" void kernel_launch(void* const* d_in, const int* in_sizes, int n_in,
                              void* d_out, int out_size, void* d_ws, size_t ws_size,
                              hipStream_t stream) {
    (void)in_sizes; (void)n_in; (void)out_size; (void)d_ws; (void)ws_size;

    const float* tf   = (const float*)d_in[0];   // target_feat [1,20,512]
    const int*   ri   = (const int*)  d_in[1];   // residue_index [1,512]
    const float* msa  = (const float*)d_in[2];   // msa_feat [1,21,512]
    const float* waw  = (const float*)d_in[3];   // w_a_w [128,20]
    const float* wab  = (const float*)d_in[4];   // w_a_b [128]
    const float* wbw  = (const float*)d_in[5];   // w_b_w [128,20]
    const float* wbb  = (const float*)d_in[6];   // w_b_b [128]
    const float* wpw  = (const float*)d_in[7];   // w_p_w [128,65]
    const float* wpb  = (const float*)d_in[8];   // w_p_b [128]
    const float* wcw  = (const float*)d_in[9];   // w_clust_w [128,256,21]
    const float* wcb  = (const float*)d_in[10];  // w_clust_b [128,256]
    const float* wmw  = (const float*)d_in[11];  // w_m_w [256,20]
    const float* wmb  = (const float*)d_in[12];  // w_m_b [256]

    float* out_msa  = (float*)d_out;                              // 256*128*512
    float* out_pair = out_msa + (size_t)CM * NC * LSEQ;           // 128*512*512

    // msa_rep: 256 c * 8 kTiles = 2048 blocks, 8 waves each
    msa_kernel<<<CM * 8, 256, 0, stream>>>(tf, msa, wcw, wcb, wmw, wmb, out_msa);

    // pair_rep: 128 c * 8 row-chunks = 1024 blocks
    pair_kernel<<<CZ * 8, 256, 0, stream>>>(tf, ri, waw, wab, wbw, wbb, wpw, wpb, out_pair);
}
